// CSSMBase_4389456577151
// MI455X (gfx1250) — matrix operand order repaired and checked by reading, not compiled, not run on hardware
//
#include <hip/hip_runtime.h>
#include <hip/hip_bf16.h>
#include <math.h>

// ---------------------------------------------------------------------------
// CSSM (2D selective SSM / Mamba block) forward for MI455X (gfx1250, wave32).
// All channel-mixing contractions run on V_WMMA_F32_16X16X32_F16.
// Activations are token-major (B, L, ch); GEMM-feeding activations are stored
// in f16 so every B fragment is ONE aligned 32-byte contiguous load and the
// GEMM inner loop is pure b128 loads + wmma (no scalar gathers, no converts).
// ---------------------------------------------------------------------------

typedef __attribute__((ext_vector_type(16))) _Float16 v16h;
typedef __attribute__((ext_vector_type(8)))  _Float16 v8h;
typedef __attribute__((ext_vector_type(8)))  float    v8f;

#define BSZ   4
#define CDIM  96
#define LSEQ  4096      // 64*64
#define DIN   192       // d_inner
#define NST   16        // d_state
#define RNK   6         // dt_rank
#define MDBL  38        // dt_rank + 2*d_state
#define LDDBL 48        // padded row stride for x_dbl

#define WMMA_F16(af, bf, acc) \
    __builtin_amdgcn_wmma_f32_16x16x32_f16(false, (af), false, (bf), (short)0, (acc), false, false)

// ---------------------------------------------------------------------------
// Pack a weight matrix A (M x K logical, f32, arbitrary strides) into the
// WMMA A-fragment layout (contiguous k per lane half, same order as B), f16, rows zero-padded to 16.
// Output: Ap[((mt*KT + kt)*32 + lane)*16 + j], KT=K/32,
//   half j -> K = kt*32 + (lane>>4)*16 + j, M row = lane&15.
// ---------------------------------------------------------------------------
__global__ __launch_bounds__(256)
void pack_a_kernel(const float* __restrict__ A, _Float16* __restrict__ Ap,
                   int M, int K, int rstride, int estride,
                   int in_zoff, int out_zoff, int total)
{
    const int z   = blockIdx.y;
    const int idx = blockIdx.x * 256 + threadIdx.x;
    if (idx >= total) return;
    const int j    = idx & 15;
    const int lane = (idx >> 4) & 31;
    const int KT   = K >> 5;
    const int kt   = (idx >> 9) % KT;
    const int mt   = idx / (KT << 9);
    const int m    = mt * 16 + (lane & 15);
    const int kk   = kt * 32 + ((lane >> 4) << 4) + j;
    const float v  = (m < M)
        ? A[(size_t)z * in_zoff + (size_t)m * rstride + (size_t)kk * estride]
        : 0.0f;
    Ap[(size_t)z * out_zoff + idx] = (_Float16)v;
}

// ---------------------------------------------------------------------------
// Transpose-convert the network input x (B, C, L) f32 -> xh (B, L, C) f16.
// ---------------------------------------------------------------------------
__global__ __launch_bounds__(256)
void pack_x_kernel(const float* __restrict__ x, _Float16* __restrict__ xh)
{
    const int idx = blockIdx.x * 256 + threadIdx.x;   // b*L*96 + l*96 + c
    if (idx >= BSZ * LSEQ * CDIM) return;
    const int c = idx % CDIM;
    const int l = (idx / CDIM) & (LSEQ - 1);
    const int b = idx / (CDIM * LSEQ);
    xh[idx] = (_Float16)x[((size_t)b * CDIM + c) * LSEQ + l];
}

// ---------------------------------------------------------------------------
// WMMA GEMM, pre-packed A, f16 token-major B:  C[b](MxN) = A(MxK) * B[b]^T,
// where Bh is (N x K) per batch (token-major activations).
// 8 waves/block; each wave computes a 16x64 strip (1 A frag -> 4 wmma/K-step).
// Stores (any subset): Cf32 token-major (ld=ldc), Ch16 token-major (ld=M),
// or Cf32 channel-major (cmajor=1, for the final output tensor).
// ---------------------------------------------------------------------------
__global__ __launch_bounds__(256)
void gemm_wmma_h(const _Float16* __restrict__ Ap,
                 const _Float16* __restrict__ Bh,
                 float* __restrict__ Cf,
                 _Float16* __restrict__ Ch,
                 int M, int K, int N, int ldc, int cmajor)
{
    const int lane = threadIdx.x & 31;
    const int wid  = threadIdx.x >> 5;
    const int n0   = (blockIdx.x * 8 + wid) * 64;
    const int mt   = blockIdx.y;
    const int b    = blockIdx.z;

    const int KT = K >> 5;
    const int nl = lane & 15;
    const int kh = (lane >> 4) * 16;   // B-fragment K half-group offset

    const _Float16* __restrict__ Abase = Ap + ((size_t)mt * KT) * 512 + lane * 16;
    const _Float16* __restrict__ Brow  =
        Bh + ((size_t)b * N + n0 + nl) * K + kh;

    v8f acc[4] = {{}, {}, {}, {}};
    for (int kt = 0; kt < KT; ++kt) {
        const v16h af = *(const v16h*)(Abase + (size_t)kt * 512);
        const int  ko = kt * 32;
        const v16h b0 = *(const v16h*)(Brow + ko);
        const v16h b1 = *(const v16h*)(Brow + (size_t)16 * K + ko);
        const v16h b2 = *(const v16h*)(Brow + (size_t)32 * K + ko);
        const v16h b3 = *(const v16h*)(Brow + (size_t)48 * K + ko);
        acc[0] = WMMA_F16(af, b0, acc[0]);
        acc[1] = WMMA_F16(af, b1, acc[1]);
        acc[2] = WMMA_F16(af, b2, acc[2]);
        acc[3] = WMMA_F16(af, b3, acc[3]);
    }

    const int mrow = mt * 16 + (lane >> 4) * 8;
    if (cmajor) {
        // channel-major f32: C[b][m][n]  (final output tensor layout)
        #pragma unroll
        for (int q = 0; q < 4; ++q) {
            const int n = n0 + q * 16 + nl;
            #pragma unroll
            for (int r = 0; r < 8; ++r)
                Cf[((size_t)b * M + mrow + r) * N + n] = acc[q][r];
        }
        return;
    }
    #pragma unroll
    for (int q = 0; q < 4; ++q) {
        const int n = n0 + q * 16 + nl;
        if (Cf) {       // token-major f32, row stride ldc (>= M, padded)
            float* __restrict__ p = Cf + ((size_t)b * N + n) * ldc + mrow;
            *(float4*)(p)     = make_float4(acc[q][0], acc[q][1], acc[q][2], acc[q][3]);
            *(float4*)(p + 4) = make_float4(acc[q][4], acc[q][5], acc[q][6], acc[q][7]);
        }
        if (Ch) {       // token-major f16, row stride M
            v8h hv;
            #pragma unroll
            for (int r = 0; r < 8; ++r) hv[r] = (_Float16)acc[q][r];
            *(v8h*)(Ch + ((size_t)b * N + n) * M + mrow) = hv;
        }
    }
}

// ---------------------------------------------------------------------------
// 3x3 conv (pad 1, 192 -> 96 ch) as 9-tap implicit GEMM on WMMA.
// B = xp1h (B, L, 192) f16; column of tap (kh,kw) at token n is token
// gy*64+gx (zero outside) -> branchless: safe row index + packed f16 mask.
// ---------------------------------------------------------------------------
__global__ __launch_bounds__(256)
void dconv_wmma_h(const _Float16* __restrict__ Apd, // 9 taps, packed
                  const _Float16* __restrict__ Bh,  // (B, 4096, 192) f16
                  _Float16* __restrict__ Ch)        // (B, 4096, 96) f16
{
    const int lane = threadIdx.x & 31;
    const int wid  = threadIdx.x >> 5;
    const int n0   = (blockIdx.x * 8 + wid) * 64;
    const int mt   = blockIdx.y;                 // 0..5 (M=96)
    const int b    = blockIdx.z;

    const int K = DIN, KT = DIN / 32;            // KT = 6
    const int tapsz = 6 * KT * 512;

    const _Float16* __restrict__ Bb = Bh + (size_t)b * LSEQ * K;
    const int nl = lane & 31 & 15;
    const int kh = (lane >> 4) * 16;

    v8f acc[4] = {{}, {}, {}, {}};

    #pragma unroll 1
    for (int tap = 0; tap < 9; ++tap) {
        const int dy = tap / 3 - 1, dx = tap % 3 - 1;
        const _Float16* Bq[4];
        _Float16 mk[4];
        #pragma unroll
        for (int q = 0; q < 4; ++q) {
            const int n  = n0 + q * 16 + nl;
            const int gy = (n >> 6) + dy;
            const int gx = (n & 63) + dx;
            const bool ok = ((unsigned)gy < 64u) & ((unsigned)gx < 64u);
            const int ns  = ok ? (gy * 64 + gx) : n;     // safe row
            mk[q] = ok ? (_Float16)1.0f : (_Float16)0.0f;
            Bq[q] = Bb + (size_t)ns * K + kh;
        }
        const _Float16* __restrict__ Abase =
            Apd + (size_t)tap * tapsz + ((size_t)mt * KT) * 512 + lane * 16;

        for (int kt = 0; kt < KT; ++kt) {
            const v16h af = *(const v16h*)(Abase + (size_t)kt * 512);
            const int  ko = kt * 32;
            v16h bf[4];
            #pragma unroll
            for (int q = 0; q < 4; ++q) {
                v16h t = *(const v16h*)(Bq[q] + ko);
                #pragma unroll
                for (int j = 0; j < 16; ++j) t[j] *= mk[q];   // packed-f16 mask
                bf[q] = t;
            }
            acc[0] = WMMA_F16(af, bf[0], acc[0]);
            acc[1] = WMMA_F16(af, bf[1], acc[1]);
            acc[2] = WMMA_F16(af, bf[2], acc[2]);
            acc[3] = WMMA_F16(af, bf[3], acc[3]);
        }
    }
    const int mrow = mt * 16 + (lane >> 4) * 8;
    #pragma unroll
    for (int q = 0; q < 4; ++q) {
        const int n = n0 + q * 16 + nl;
        v8h hv;
        #pragma unroll
        for (int r = 0; r < 8; ++r) hv[r] = (_Float16)acc[q][r];
        *(v8h*)(Ch + ((size_t)b * LSEQ + n) * CDIM + mrow) = hv;
    }
}

// ---------------------------------------------------------------------------
// Causal depthwise conv1d (k=4, left pad 3) + bias + SiLU; token-major.
// Writes both f32 (for the scan's u) and f16 (x_proj GEMM B operand).
// ---------------------------------------------------------------------------
__global__ __launch_bounds__(256)
void conv1d_silu_kernel(const float* __restrict__ xzt,  // (B, L, 384) f32
                        const float* __restrict__ w,    // (192,4)
                        const float* __restrict__ bias, // (192)
                        float* __restrict__ xct,        // (B, L, 192) f32
                        _Float16* __restrict__ xch)     // (B, L, 192) f16
{
    const int idx = blockIdx.x * 256 + threadIdx.x;     // b*L*192 + t*192 + d
    if (idx >= BSZ * LSEQ * DIN) return;
    const int d = idx % DIN;
    const int t = (idx / DIN) & (LSEQ - 1);
    const int b = idx / (DIN * LSEQ);
    const float* __restrict__ xb = xzt + (size_t)b * LSEQ * (2 * DIN);
    float s = bias[d];
    #pragma unroll
    for (int k = 0; k < 4; ++k) {
        const int tt = t - 3 + k;
        if (tt >= 0) s = fmaf(w[d * 4 + k], xb[(size_t)tt * (2 * DIN) + d], s);
    }
    const float v = s / (1.0f + __expf(-s));            // SiLU
    xct[idx] = v;
    xch[idx] = (_Float16)v;
}

// ---------------------------------------------------------------------------
// delta = softplus(dt @ dt_proj_w^T + dt_proj_b); dt = x_dbl cols [0,6).
// ---------------------------------------------------------------------------
__global__ __launch_bounds__(256)
void delta_kernel(const float* __restrict__ xdblt, // (B, L, 48) f32
                  const float* __restrict__ dtw,   // (192,6)
                  const float* __restrict__ dtb,   // (192)
                  float* __restrict__ deltat)      // (B, L, 192) f32
{
    const int idx = blockIdx.x * 256 + threadIdx.x;
    if (idx >= BSZ * LSEQ * DIN) return;
    const int d = idx % DIN;
    const int t = (idx / DIN) & (LSEQ - 1);
    const int b = idx / (DIN * LSEQ);
    const float* __restrict__ xr = xdblt + ((size_t)b * LSEQ + t) * LDDBL;
    float s = dtb[d];
    #pragma unroll
    for (int r = 0; r < RNK; ++r)
        s = fmaf(dtw[d * RNK + r], xr[r], s);
    deltat[idx] = (s > 20.0f) ? s : log1pf(__expf(s));
}

// ---------------------------------------------------------------------------
// Selective scan (sequential in L): one wave per (b,d), lane n = state n
// (duplicated across half-waves).  Token-major layout makes Bc/Cc one
// contiguous 16-lane load per step.  Fuses y = sum_n h*C + D*u and the
// SiLU(z) gate; emits f16 directly (out_proj GEMM B operand).
// ---------------------------------------------------------------------------
__global__ __launch_bounds__(256)
void scan_kernel(const float* __restrict__ deltat, // (B, L, 192)
                 const float* __restrict__ xct,    // (B, L, 192) (= u)
                 const float* __restrict__ xdblt,  // (B, L, 48): 6..21 Bc, 22..37 Cc
                 const float* __restrict__ xzt,    // (B, L, 384): cols 192.. = z
                 const float* __restrict__ A_log,  // (192,16)
                 const float* __restrict__ Dvec,   // (192)
                 _Float16* __restrict__ ygh)       // (B, L, 192) f16
{
    const int wave = (blockIdx.x * blockDim.x + threadIdx.x) >> 5;
    const int lane = threadIdx.x & 31;
    if (wave >= BSZ * DIN) return;
    const int b = wave / DIN, d = wave % DIN;
    const int n = lane & 15;

    const float An = -__expf(A_log[d * NST + n]);
    const float Dd = Dvec[d];

    const float* __restrict__ dl = deltat + (size_t)b * LSEQ * DIN + d;
    const float* __restrict__ ul = xct    + (size_t)b * LSEQ * DIN + d;
    const float* __restrict__ xd = xdblt  + (size_t)b * LSEQ * LDDBL;
    const float* __restrict__ zl = xzt    + (size_t)b * LSEQ * (2 * DIN) + DIN + d;
    _Float16* __restrict__    yl = ygh    + (size_t)b * LSEQ * DIN + d;

    float h = 0.0f;
    #pragma unroll 2
    for (int t = 0; t < LSEQ; ++t) {
        const float dt = dl[(size_t)t * DIN];             // broadcast
        const float u  = ul[(size_t)t * DIN];             // broadcast
        const float Bn = xd[(size_t)t * LDDBL + RNK + n];       // 16-lane coalesced
        const float Cn = xd[(size_t)t * LDDBL + RNK + NST + n]; // 16-lane coalesced
        const float z  = zl[(size_t)t * (2 * DIN)];       // broadcast

        h = __expf(dt * An) * h + dt * Bn * u;
        float acc = h * Cn;
        acc += __shfl_xor(acc, 1, 16);
        acc += __shfl_xor(acc, 2, 16);
        acc += __shfl_xor(acc, 4, 16);
        acc += __shfl_xor(acc, 8, 16);
        if (lane == 0) {
            const float y = acc + Dd * u;
            yl[(size_t)t * DIN] = (_Float16)(y * (z / (1.0f + __expf(-z))));
        }
    }
}

// ---------------------------------------------------------------------------
// Host-side orchestration
// ---------------------------------------------------------------------------
extern "C" void kernel_launch(void* const* d_in, const int* in_sizes, int n_in,
                              void* d_out, int out_size, void* d_ws, size_t ws_size,
                              hipStream_t stream)
{
    const float* x        = (const float*)d_in[0];   // (4,96,64,64)
    const float* proj_w   = (const float*)d_in[1];   // (192,96,1,1)
    const float* dconv_w  = (const float*)d_in[2];   // (96,192,3,3)
    const float* inproj_w = (const float*)d_in[3];   // (384,96)
    const float* conv1d_w = (const float*)d_in[4];   // (192,4)
    const float* conv1d_b = (const float*)d_in[5];   // (192)
    const float* xproj_w  = (const float*)d_in[6];   // (38,192)
    const float* dtproj_w = (const float*)d_in[7];   // (192,6)
    const float* dtproj_b = (const float*)d_in[8];   // (192)
    const float* A_log    = (const float*)d_in[9];   // (192,16)
    const float* Dvec     = (const float*)d_in[10];  // (192)
    const float* outp_w   = (const float*)d_in[11];  // (96,192)
    float* out = (float*)d_out;                      // (4,96,64,64)

    // ---- workspace: f32 token-major activations, then f16 buffers ----
    float* ws = (float*)d_ws;
    float* xzt    = ws;                                         // (4,4096,384)
    float* xct    = xzt    + (size_t)BSZ * LSEQ * 2 * DIN;      // (4,4096,192)
    float* xdblt  = xct    + (size_t)BSZ * LSEQ * DIN;          // (4,4096, 48)
    float* deltat = xdblt  + (size_t)BSZ * LSEQ * LDDBL;        // (4,4096,192)
    _Float16* h16 = (_Float16*)(deltat + (size_t)BSZ * LSEQ * DIN);
    _Float16* xh   = h16;                                       // (4,4096, 96)
    _Float16* xp1h = xh   + (size_t)BSZ * LSEQ * CDIM;          // (4,4096,192)
    _Float16* xp2h = xp1h + (size_t)BSZ * LSEQ * DIN;           // (4,4096, 96)
    _Float16* xch  = xp2h + (size_t)BSZ * LSEQ * CDIM;          // (4,4096,192)
    _Float16* ygh  = xch  + (size_t)BSZ * LSEQ * DIN;           // (4,4096,192)
    _Float16* pk   = ygh  + (size_t)BSZ * LSEQ * DIN;

    const int szProj = 12 * 3 * 512;   // Mtiles * KT * 512
    const int szInp  = 24 * 3 * 512;
    const int szXp   =  3 * 6 * 512;
    const int szOutp =  6 * 6 * 512;
    const int szDcT  =  6 * 6 * 512;   // per tap
    _Float16* pProj = pk;
    _Float16* pInp  = pProj + szProj;
    _Float16* pXp   = pInp  + szInp;
    _Float16* pOutp = pXp   + szXp;
    _Float16* pDc   = pOutp + szOutp;  // 9 taps

    // ---- 0) pack weights + transpose-convert input (all tiny) ----
    pack_a_kernel<<<dim3((szProj + 255) / 256, 1), 256, 0, stream>>>(
        proj_w,   pProj, DIN,     CDIM, CDIM,    1, 0, 0, szProj);
    pack_a_kernel<<<dim3((szInp  + 255) / 256, 1), 256, 0, stream>>>(
        inproj_w, pInp,  2 * DIN, CDIM, CDIM,    1, 0, 0, szInp);
    pack_a_kernel<<<dim3((szXp   + 255) / 256, 1), 256, 0, stream>>>(
        xproj_w,  pXp,   MDBL,    DIN,  DIN,     1, 0, 0, szXp);
    pack_a_kernel<<<dim3((szOutp + 255) / 256, 1), 256, 0, stream>>>(
        outp_w,   pOutp, CDIM,    DIN,  DIN,     1, 0, 0, szOutp);
    pack_a_kernel<<<dim3((szDcT  + 255) / 256, 9), 256, 0, stream>>>(
        dconv_w,  pDc,   CDIM,    DIN,  DIN * 9, 9, 1, szDcT, szDcT);
    {
        const int total = BSZ * LSEQ * CDIM;
        pack_x_kernel<<<(total + 255) / 256, 256, 0, stream>>>(x, xh);
    }

    const int GX = LSEQ / 512;   // 8 wave-strips of 64 tokens

    // 1) proj 1x1: (192x96) x (96x4096) -> xp1h (f16 only; feeds dconv)
    gemm_wmma_h<<<dim3(GX, DIN / 16, BSZ), 256, 0, stream>>>(
        pProj, xh, nullptr, xp1h, DIN, CDIM, LSEQ, 0, 0);

    // 2) 3x3 conv as 9-tap implicit WMMA GEMM -> xp2h (f16 only)
    dconv_wmma_h<<<dim3(GX, CDIM / 16, BSZ), 256, 0, stream>>>(
        pDc, xp1h, xp2h);

    // 3) in_proj: (384x96) x (96x4096) -> xzt (f32; feeds conv1d + gate)
    gemm_wmma_h<<<dim3(GX, (2 * DIN) / 16, BSZ), 256, 0, stream>>>(
        pInp, xp2h, xzt, nullptr, 2 * DIN, CDIM, LSEQ, 2 * DIN, 0);

    // 4) causal depthwise conv1d + SiLU -> xct (f32) + xch (f16)
    {
        const int total = BSZ * LSEQ * DIN;
        conv1d_silu_kernel<<<(total + 255) / 256, 256, 0, stream>>>(
            xzt, conv1d_w, conv1d_b, xct, xch);
    }

    // 5) x_proj: (38x192) x (192x4096) -> xdblt (f32, ld=48 padded)
    gemm_wmma_h<<<dim3(GX, 3, BSZ), 256, 0, stream>>>(
        pXp, xch, xdblt, nullptr, LDDBL, DIN, LSEQ, LDDBL, 0);

    // 6) delta = softplus(dt @ dt_proj^T + b)
    {
        const int total = BSZ * LSEQ * DIN;
        delta_kernel<<<(total + 255) / 256, 256, 0, stream>>>(
            xdblt, dtproj_w, dtproj_b, deltat);
    }

    // 7) selective scan + skip + gate -> ygh (f16; feeds out_proj)
    {
        const int waves = BSZ * DIN;
        scan_kernel<<<(waves * 32 + 255) / 256, 256, 0, stream>>>(
            deltat, xct, xdblt, xzt, A_log, Dvec, ygh);
    }

    // 8) out_proj: (96x192) x (192x4096) -> d_out, channel-major (B,C,H,W)
    gemm_wmma_h<<<dim3(GX, CDIM / 16, BSZ), 256, 0, stream>>>(
        pOutp, ygh, out, nullptr, CDIM, DIN, LSEQ, 0, 1);
}
